// SSM_74938589381379
// MI455X (gfx1250) — matrix operand order repaired and checked by reading, not compiled, not run on hardware
//
#include <hip/hip_runtime.h>

typedef __attribute__((ext_vector_type(16))) __bf16 v16b;
typedef __attribute__((ext_vector_type(8)))  float  v8f;

#define BSZ   4
#define LSEQ  4096
#define DDIM  1024
#define SDIM  64
#define TC    16
#define NCH   (LSEQ / TC)      /* 256 chunks per batch   */
#define CROWS (BSZ * NCH)      /* 1024 chunk-rows        */
#define ROWS  (BSZ * LSEQ)     /* 16384 time rows        */

// ---------------------------------------------------------------------------
// gfx1250 async global->LDS copies (ASYNCcnt-tracked, no VGPR round-trip).
// Inline asm per CDNA5_HIP.md (portable across ROCm7.2 / clang-23 toolchains).
// ---------------------------------------------------------------------------
__device__ __forceinline__ void async_b128(uint32_t lds_off, const void* g) {
  asm volatile("global_load_async_to_lds_b128 %0, %1, off"
               :: "v"(lds_off), "v"((unsigned long long)(uintptr_t)g)
               : "memory");
}
__device__ __forceinline__ void async_b64(uint32_t lds_off, const void* g) {
  asm volatile("global_load_async_to_lds_b64 %0, %1, off"
               :: "v"(lds_off), "v"((unsigned long long)(uintptr_t)g)
               : "memory");
}
__device__ __forceinline__ void wait_async0() {
  asm volatile("s_wait_asynccnt 0" ::: "memory");
}
__device__ __forceinline__ uint32_t lds_off_of(const void* p) {
  return (uint32_t)(uintptr_t)p;
}

// ---------------------------------------------------------------------------
// WMMA fragment loaders (bf16, wave32, 16x16x32)
// A layout (16-bit A 16x32): lane m=lane&15, half=lane>>4,
//   element i -> k = ((i>>3)<<4) + half*8 + (i&7)
// B layout (16-bit B 32x16, n-major staged LDS Bt[n][k]):
//   lane n=lane&15, half=lane>>4, element i -> k = ((i>>3)<<4) + half*8 + (i&7)
// C/D layout: lane n=lane&15, VGPR r -> m = r + (lane>=16 ? 8 : 0)
// ---------------------------------------------------------------------------
__device__ __forceinline__ v16b load_a_frag(const __bf16* lds, int pitch,
                                            int rowBase, int kBase, int lane) {
  const int m = lane & 15, half = lane >> 4;
  const __bf16* p = lds + (rowBase + m) * pitch + kBase;
  v16b a;
#pragma unroll
  for (int i = 0; i < 16; ++i) {
    int k = ((i >> 3) << 4) + half * 8 + (i & 7);
    a[i] = p[k];
  }
  return a;
}

__device__ __forceinline__ v16b load_b_frag(const __bf16* ldsBt, int pitch,
                                            int nBase, int kBase, int lane) {
  const int n = lane & 15, half = lane >> 4;
  const __bf16* p = ldsBt + (nBase + n) * pitch + kBase;
  v16b b;
#pragma unroll
  for (int i = 0; i < 16; ++i) b[i] = p[((i >> 3) << 4) + half * 8 + (i & 7)];
  return b;
}

// ---------------------------------------------------------------------------
// K1: uB = x @ Bm^T  -> stored as bf16 [16384 x 64]
// (bit-identical to the chunk-row matrix U [1024 x 1024])
// block: 256 thr / 8 waves, 128 rows x 64 cols, K-tile 32.
// fp32 tiles are converted to packed bf16 (v_cvt_pk_bf16_f32) and staged with
// b128 LDS stores; epilogue transposes through LDS for b128 global stores.
// ---------------------------------------------------------------------------
__global__ void k1_uB(const float* __restrict__ x, const float* __restrict__ Bm,
                      __bf16* __restrict__ Ubf) {
  __shared__ __align__(16) __bf16 lx[128 * 64];   // input: 128x32; epilogue: 128x64
  __shared__ __align__(16) __bf16 lb[64 * 32];
  const int t = threadIdx.x, lane = t & 31, wave = t >> 5;
  const int rowBase = blockIdx.x * 128;
  v8f acc[4] = {};
  const int xr = t >> 1, xc = (t & 1) * 16;       // x stage: 16 elems/thread
  const int br = t >> 2, bc = (t & 3) * 8;        // Bm stage: 8 elems/thread
  for (int k0 = 0; k0 < DDIM; k0 += 32) {
    {
      union { __bf16 h[16]; uint4 q[2]; } u;
#pragma unroll
      for (int e = 0; e < 4; ++e) {
        float4 f = *(const float4*)&x[(size_t)(rowBase + xr) * DDIM + k0 + xc + e * 4];
        u.h[e * 4 + 0] = (__bf16)f.x; u.h[e * 4 + 1] = (__bf16)f.y;
        u.h[e * 4 + 2] = (__bf16)f.z; u.h[e * 4 + 3] = (__bf16)f.w;
      }
      *(uint4*)&lx[xr * 32 + xc]     = u.q[0];
      *(uint4*)&lx[xr * 32 + xc + 8] = u.q[1];
    }
    {
      union { __bf16 h[8]; uint4 q; } u;
#pragma unroll
      for (int e = 0; e < 2; ++e) {
        float4 f = *(const float4*)&Bm[(size_t)br * DDIM + k0 + bc + e * 4];
        u.h[e * 4 + 0] = (__bf16)f.x; u.h[e * 4 + 1] = (__bf16)f.y;
        u.h[e * 4 + 2] = (__bf16)f.z; u.h[e * 4 + 3] = (__bf16)f.w;
      }
      *(uint4*)&lb[br * 32 + bc] = u.q;
    }
    __syncthreads();
    v16b a = load_a_frag(lx, 32, wave * 16, 0, lane);
#pragma unroll
    for (int nt = 0; nt < 4; ++nt) {
      v16b b = load_b_frag(lb, 32, nt * 16, 0, lane);
      acc[nt] = __builtin_amdgcn_wmma_f32_16x16x32_bf16(
          false, a, false, b, (short)0, acc[nt], false, false);
    }
    __syncthreads();
    if (k0 + 32 < DDIM)
      __builtin_prefetch(&x[(size_t)(rowBase + xr) * DDIM + k0 + 32 + xc], 0, 0);
  }
  // epilogue: accs -> LDS (pitch 64) -> coalesced b128 global stores
  const int n = lane & 15, halfm = (lane >> 4) * 8;
#pragma unroll
  for (int nt = 0; nt < 4; ++nt)
#pragma unroll
    for (int r = 0; r < 8; ++r)
      lx[(wave * 16 + halfm + r) * 64 + nt * 16 + n] = (__bf16)acc[nt][r];
  __syncthreads();
  {
    const int r = t >> 1, c = (t & 1) * 32;
#pragma unroll
    for (int e = 0; e < 4; ++e)
      *(uint4*)&Ubf[(size_t)(rowBase + r) * 64 + c + e * 8] =
          *(const uint4*)&lx[r * 64 + c + e * 8];
  }
}

// ---------------------------------------------------------------------------
// K2: powers P_p = (A^T)^p, p = 0..16, fp32, single block
// P_p[j][s] = sum_q P_{p-1}[j][q] * A[s][q]
// ---------------------------------------------------------------------------
__global__ void k2_powers(const float* __restrict__ A, float* __restrict__ P) {
  __shared__ float prev[SDIM * SDIM];
  const int t = threadIdx.x;
#pragma unroll
  for (int e = 0; e < 16; ++e) {
    int idx = e * 256 + t;
    int j = idx >> 6, s = idx & 63;
    float v = (j == s) ? 1.f : 0.f;
    P[idx] = v; prev[idx] = v;
  }
  __syncthreads();
  for (int p = 1; p <= TC; ++p) {
    float regs[16];
#pragma unroll
    for (int e = 0; e < 16; ++e) {
      int idx = e * 256 + t;
      int j = idx >> 6, s = idx & 63;
      float a = 0.f;
      for (int q = 0; q < SDIM; ++q) a += prev[j * 64 + q] * A[s * 64 + q];
      regs[e] = a;
    }
    __syncthreads();
#pragma unroll
    for (int e = 0; e < 16; ++e) {
      int idx = e * 256 + t;
      prev[idx] = regs[e];
      P[p * 4096 + idx] = regs[e];
    }
    __syncthreads();
  }
}

// ---------------------------------------------------------------------------
// K3: build bf16 operand matrices
//  Gt   [1024 n][1024 k] : Gt[t*64+s][kk*64+j] = (t>=kk) ? P_{t-kk}[j][s] : 0
//  PcatT[1024 n][64 k]   : PcatT[t*64+s][j]    = P_{t+1}[j][s]
//  Cbf  [1024 d][64 s]   : bf16(C)   (already n-major for GEMM2)
// ---------------------------------------------------------------------------
__global__ void k3_build(const float* __restrict__ P, const float* __restrict__ C,
                         __bf16* __restrict__ Gt, __bf16* __restrict__ PcatT,
                         __bf16* __restrict__ Cbf) {
  const int NG = 1024 * 1024, NP = 1024 * 64;
  int i = blockIdx.x * 256 + threadIdx.x;
  if (i < NG) {
    int nn = i >> 10, k = i & 1023;
    int tt = nn >> 6, s = nn & 63;
    int kk = k >> 6, j = k & 63;
    float v = (tt >= kk) ? P[(tt - kk) * 4096 + j * 64 + s] : 0.f;
    Gt[i] = (__bf16)v;
  } else if (i < NG + NP) {
    int ii = i - NG;
    int nn = ii >> 6, j = ii & 63;
    int tt = nn >> 6, s = nn & 63;
    PcatT[ii] = (__bf16)P[(tt + 1) * 4096 + j * 64 + s];
  } else if (i < NG + 2 * NP) {
    int ii = i - NG - NP;
    Cbf[ii] = (__bf16)C[ii];
  }
}

// ---------------------------------------------------------------------------
// K4: Sintra = U @ G   (1024 x 1024 x 1024, bf16 WMMA, fp32 out)
// block 64x64 tile, 8 waves (4 row-tiles x 2 col-pairs), K-tile 32.
// Operands staged with GLOBAL_LOAD_ASYNC_TO_LDS_B128 (ASYNCcnt).
// ---------------------------------------------------------------------------
__global__ void k4_sintra(const __bf16* __restrict__ Ubf,
                          const __bf16* __restrict__ Gt,
                          float* __restrict__ Sintra) {
  __shared__ __align__(16) __bf16 la[64 * 32];
  __shared__ __align__(16) __bf16 lbt[64 * 32];
  const int t = threadIdx.x, lane = t & 31, wave = t >> 5;
  const int by = blockIdx.x >> 4, bx = blockIdx.x & 15;
  const int rowBase = by * 64, nBase = bx * 64;
  const int mB = (wave >> 1) * 16, nB = (wave & 1) * 32;
  const int r = t >> 2, coff = (t & 3) * 8;
  const uint32_t laOff  = lds_off_of(&la[r * 32 + coff]);
  const uint32_t lbtOff = lds_off_of(&lbt[r * 32 + coff]);
  v8f acc[2] = {};
  for (int k0 = 0; k0 < 1024; k0 += 32) {
    async_b128(laOff,  &Ubf[(size_t)(rowBase + r) * 1024 + k0 + coff]);
    async_b128(lbtOff, &Gt[(size_t)(nBase + r) * 1024 + k0 + coff]);
    wait_async0();
    __syncthreads();
    v16b a = load_a_frag(la, 32, mB, 0, lane);
#pragma unroll
    for (int j = 0; j < 2; ++j) {
      v16b b = load_b_frag(lbt, 32, nB + j * 16, 0, lane);
      acc[j] = __builtin_amdgcn_wmma_f32_16x16x32_bf16(
          false, a, false, b, (short)0, acc[j], false, false);
    }
    __syncthreads();
  }
  const int n = lane & 15, halfm = (lane >> 4) * 8;
#pragma unroll
  for (int j = 0; j < 2; ++j)
#pragma unroll
    for (int rr = 0; rr < 8; ++rr)
      Sintra[(size_t)(rowBase + mB + halfm + rr) * 1024 + nBase + nB + j * 16 + n] =
          acc[j][rr];
}

// ---------------------------------------------------------------------------
// K5: sequential chunk-carry scan (only 256 dependent steps, 4 batches in par)
//  h_{c+1} = h_c @ P16 + Sintra[chunk c, t=15, :] ; Hbf stores h into chunk c
// ---------------------------------------------------------------------------
__global__ void k5_scan(const float* __restrict__ P,
                        const float* __restrict__ Sintra,
                        __bf16* __restrict__ Hbf) {
  __shared__ float lp[SDIM * SDIM];
  __shared__ float lh[BSZ * SDIM];
  const int t = threadIdx.x;
#pragma unroll
  for (int e = 0; e < 16; ++e) lp[e * 256 + t] = P[TC * 4096 + e * 256 + t];
  lh[t] = 0.f;
  __syncthreads();
  const int b = t >> 6, s = t & 63;
  for (int c = 0; c < NCH; ++c) {
    const int chrow = b * NCH + c;
    Hbf[chrow * 64 + s] = (__bf16)lh[b * 64 + s];
    float acc = Sintra[(size_t)chrow * 1024 + (TC - 1) * 64 + s];
    for (int j = 0; j < 64; ++j) acc += lh[b * 64 + j] * lp[j * 64 + s];
    __syncthreads();
    lh[b * 64 + s] = acc;
    __syncthreads();
  }
}

// ---------------------------------------------------------------------------
// K6: states = Sintra + H @ Pcat  (1024 x 1024, K=64), bf16 out
// Sintra is fed as the WMMA C accumulator; operands staged async.
// ---------------------------------------------------------------------------
__global__ void k6_states(const __bf16* __restrict__ Hbf,
                          const __bf16* __restrict__ PcatT,
                          const float* __restrict__ Sintra,
                          __bf16* __restrict__ Stbf) {
  __shared__ __align__(16) __bf16 la[64 * 64];
  __shared__ __align__(16) __bf16 lbt[64 * 64];
  const int t = threadIdx.x, lane = t & 31, wave = t >> 5;
  const int by = blockIdx.x >> 4, bx = blockIdx.x & 15;
  const int rowBase = by * 64, nBase = bx * 64;
  const int r = t >> 2, coff = (t & 3) * 16;
#pragma unroll
  for (int u = 0; u < 2; ++u) {
    async_b128(lds_off_of(&la[r * 64 + coff + u * 8]),
               &Hbf[(size_t)(rowBase + r) * 64 + coff + u * 8]);
    async_b128(lds_off_of(&lbt[r * 64 + coff + u * 8]),
               &PcatT[(size_t)(nBase + r) * 64 + coff + u * 8]);
  }
  wait_async0();
  __syncthreads();
  const int mB = (wave >> 1) * 16, nB = (wave & 1) * 32;
  const int n = lane & 15, halfm = (lane >> 4) * 8;
#pragma unroll
  for (int j = 0; j < 2; ++j) {
    const int ncol = nBase + nB + j * 16 + n;
    v8f acc;
#pragma unroll
    for (int rr = 0; rr < 8; ++rr)
      acc[rr] = Sintra[(size_t)(rowBase + mB + halfm + rr) * 1024 + ncol];
#pragma unroll
    for (int kk = 0; kk < 64; kk += 32) {
      v16b a = load_a_frag(la, 64, mB, kk, lane);
      v16b b = load_b_frag(lbt, 64, nB + j * 16, kk, lane);
      acc = __builtin_amdgcn_wmma_f32_16x16x32_bf16(
          false, a, false, b, (short)0, acc, false, false);
    }
#pragma unroll
    for (int rr = 0; rr < 8; ++rr)
      Stbf[(size_t)(rowBase + mB + halfm + rr) * 1024 + ncol] = (__bf16)acc[rr];
  }
}

// ---------------------------------------------------------------------------
// K7: fused y = states @ C^T -> exact-erf GELU -> LayerNorm -> nan_to_num
// block owns 16 full rows (D=1024) so y never touches HBM.
// dynamic LDS: act 64KB + C slice 16KB + states 2KB + reduce 2.3KB
// C/states tiles staged with async-to-LDS.
// ---------------------------------------------------------------------------
__global__ void k7_out(const __bf16* __restrict__ Stbf,
                       const __bf16* __restrict__ Cbf,
                       const float* __restrict__ gamma,
                       const float* __restrict__ beta,
                       float* __restrict__ out) {
  extern __shared__ __align__(16) char smem[];
  float*  lact = (float*)smem;                          // 16*1024 f32
  __bf16* lc   = (__bf16*)(smem + 65536);               // 128*64
  __bf16* lst  = (__bf16*)(smem + 65536 + 16384);       // 16*64
  float*  lred = (float*)(smem + 65536 + 16384 + 2048); // 576 f32
  const int t = threadIdx.x, lane = t & 31, wave = t >> 5;
  const int rowBase = blockIdx.x * 16;
  {                                                     // stage states 16x64
    int idx = t * 4;
    int rr = idx >> 6, cc = idx & 63;
    async_b64(lds_off_of(&lst[idx]), &Stbf[(size_t)(rowBase + rr) * 64 + cc]);
  }
  const int n = lane & 15, halfm = (lane >> 4) * 8;
  const int crr = t >> 1, ccc = (t & 1) * 32;
  for (int nc = 0; nc < 8; ++nc) {
    const int nChunk = nc * 128;
#pragma unroll
    for (int u = 0; u < 4; ++u)                         // stage C slice 128x64
      async_b128(lds_off_of(&lc[crr * 64 + ccc + u * 8]),
                 &Cbf[(size_t)(nChunk + crr) * 64 + ccc + u * 8]);
    wait_async0();
    __syncthreads();
    v8f acc = {};
#pragma unroll
    for (int kk = 0; kk < 64; kk += 32) {
      v16b a = load_a_frag(lst, 64, 0, kk, lane);
      v16b b = load_b_frag(lc, 64, wave * 16, kk, lane);
      acc = __builtin_amdgcn_wmma_f32_16x16x32_bf16(
          false, a, false, b, (short)0, acc, false, false);
    }
    const int col = nChunk + wave * 16 + n;
#pragma unroll
    for (int rr = 0; rr < 8; ++rr) {
      float v = acc[rr];
      float g = 0.5f * v * (1.f + erff(v * 0.70710678118654752f));
      lact[(halfm + rr) * 1024 + col] = g;
    }
    __syncthreads();
  }
  // LayerNorm over D=1024 per row
  const int row = t >> 4, l16 = t & 15;
  float s1 = 0.f, s2 = 0.f;
  for (int c = l16; c < 1024; c += 16) {
    float v = lact[row * 1024 + c];
    s1 += v; s2 += v * v;
  }
  lred[row * 16 + l16] = s1;
  lred[256 + row * 16 + l16] = s2;
  __syncthreads();
  if (t < 16) {
    float a1 = 0.f, a2 = 0.f;
    for (int u = 0; u < 16; ++u) {
      a1 += lred[t * 16 + u];
      a2 += lred[256 + t * 16 + u];
    }
    float mean = a1 * (1.f / 1024.f);
    float var  = a2 * (1.f / 1024.f) - mean * mean;
    lred[512 + t * 2]     = mean;
    lred[512 + t * 2 + 1] = rsqrtf(var + 1e-5f);
  }
  __syncthreads();
  const float mean = lred[512 + row * 2];
  const float rstd = lred[512 + row * 2 + 1];
  for (int c = l16; c < 1024; c += 16) {
    float v = (lact[row * 1024 + c] - mean) * rstd * gamma[c] + beta[c];
    if (v != v) v = 0.f;
    v = fminf(fmaxf(v, -1000000.0f), 1000000.0f);
    out[(size_t)(rowBase + row) * 1024 + c] = v;
  }
}

// ---------------------------------------------------------------------------
extern "C" void kernel_launch(void* const* d_in, const int* in_sizes, int n_in,
                              void* d_out, int out_size, void* d_ws, size_t ws_size,
                              hipStream_t stream) {
  const float* x     = (const float*)d_in[0];
  const float* A     = (const float*)d_in[1];
  const float* Bm    = (const float*)d_in[2];
  const float* C     = (const float*)d_in[3];
  const float* gamma = (const float*)d_in[4];
  const float* beta  = (const float*)d_in[5];
  float* out = (float*)d_out;

  // workspace carve (~10.7 MB total)
  char* ws = (char*)d_ws;
  __bf16* Ubf    = (__bf16*)(ws);                 // 16384*64*2   = 2 MB
  float*  P      = (float*) (ws + 2097152);       // 17*4096*4    = 272 KB
  __bf16* Gt     = (__bf16*)(ws + 2375680);       // 1024*1024*2  = 2 MB
  __bf16* PcatT  = (__bf16*)(ws + 4472832);       // 1024*64*2    = 128 KB
  __bf16* Cbf    = (__bf16*)(ws + 4603904);       // 1024*64*2    = 128 KB
  float*  Sintra = (float*) (ws + 4734976);       // 1024*1024*4  = 4 MB
  __bf16* Hbf    = (__bf16*)(ws + 8929280);       // 1024*64*2    = 128 KB
  __bf16* Stbf   = (__bf16*)(ws + 9060352);       // 16384*64*2   = 2 MB

  k1_uB    <<<ROWS / 128, 256, 0, stream>>>(x, Bm, Ubf);
  k2_powers<<<1,          256, 0, stream>>>(A, P);
  k3_build <<<(1024 * 1024 + 2 * 1024 * 64) / 256, 256, 0, stream>>>(P, C, Gt, PcatT, Cbf);
  k4_sintra<<<256,        256, 0, stream>>>(Ubf, Gt, Sintra);
  k5_scan  <<<1,          256, 0, stream>>>(P, Sintra, Hbf);
  k6_states<<<256,        256, 0, stream>>>(Hbf, PcatT, Sintra, Stbf);
  k7_out   <<<ROWS / 16,  256, 86272, stream>>>(Stbf, Cbf, gamma, beta, out);
}
